// DualGatedCrossAttentionAdapter_58729382806133
// MI455X (gfx1250) — matrix operand order repaired and checked by reading, not compiled, not run on hardware
//
#include <hip/hip_runtime.h>
#include <math.h>

// ---------------------------------------------------------------------------
// Types & WMMA fragment helpers (CDNA5 wave32, V_WMMA_F32_16X16X32_BF16)
// ---------------------------------------------------------------------------
typedef __bf16 bf16_t;
typedef __attribute__((ext_vector_type(16))) __bf16 v16bf;
typedef __attribute__((ext_vector_type(8)))  float  v8f;

struct __align__(16) U128 { unsigned x, y, z, w; };

union FragU { v16bf v; U128 q[2]; };

__device__ __forceinline__ bf16_t f2bf(float f) {
  union { float f; unsigned u; } a; a.f = f;
  unsigned r = a.u + 0x7fffu + ((a.u >> 16) & 1u);   // round-to-nearest-even
  union { unsigned short s; bf16_t b; } o; o.s = (unsigned short)(r >> 16);
  return o.b;
}
__device__ __forceinline__ float bf2f(bf16_t x) {
  union { bf16_t b; unsigned short s; } u; u.b = x;
  union { unsigned i; float f; } o; o.i = ((unsigned)u.s) << 16;
  return o.f;
}

// A-matrix 16x32 bf16 fragment (ISA 7.12.2): lane m=lane&15;
// lanes<16 hold K {kb+0..7, kb+16..23}, lanes>=16 hold K {kb+8..15, kb+24..31}.
__device__ __forceinline__ v16bf ld_a(const bf16_t* base, int stride, int lane) {
  const bf16_t* p = base + (lane & 15) * stride + ((lane >> 4) << 3);
  FragU u;
  u.q[0] = *(const U128*)(p);
  u.q[1] = *(const U128*)(p + 16);
  return u.v;
}
// B-matrix 32x16 bf16 fragment: lane col n=lane&15; lanes<16 hold K {kb+0..7, kb+16..23},
// lanes>=16 hold K {kb+8..15, kb+24..31} (matches A-fragment k order).
__device__ __forceinline__ v16bf ld_b(const bf16_t* base, int stride, int lane) {
  const bf16_t* p = base + (lane & 15) * stride + ((lane >> 4) << 3);
  FragU u;
  u.q[0] = *(const U128*)(p);
  u.q[1] = *(const U128*)(p + 16);
  return u.v;
}
__device__ __forceinline__ v8f wmma_bf(v16bf a, v16bf b, v8f c) {
  return __builtin_amdgcn_wmma_f32_16x16x32_bf16(false, a, false, b, (short)0, c,
                                                 false, false);
}

// CDNA5 async global->LDS copy (16B per lane), tracked by ASYNCcnt.
__device__ __forceinline__ void async_copy_b128(const void* lds_dst, const void* gsrc) {
  // generic pointer to LDS: low 32 bits are the LDS offset (aperture in high bits)
  unsigned off = (unsigned)(unsigned long long)lds_dst;
  asm volatile("global_load_async_to_lds_b128 %0, %1, off"
               :: "v"(off), "v"(gsrc) : "memory");
}
__device__ __forceinline__ void wait_async() {
  asm volatile("s_wait_asynccnt 0x0" ::: "memory");
}

// ---------------------------------------------------------------------------
// Problem constants
// ---------------------------------------------------------------------------
#define BB 4
#define LL 2048
#define NN 256
#define DD 2560
#define HH 8
#define HD 320
#define RR 64

// ---------------------------------------------------------------------------
// fp32 -> bf16 conversion
// ---------------------------------------------------------------------------
__global__ __launch_bounds__(256) void k_cvt(const float* __restrict__ s,
                                             bf16_t* __restrict__ d, int n) {
  int i = blockIdx.x * 256 + threadIdx.x;
  if (i < n) d[i] = f2bf(s[i]);
}

// ---------------------------------------------------------------------------
// RMS norm: one block per row of text_hidden, writes bf16 x_norm
// ---------------------------------------------------------------------------
__global__ __launch_bounds__(256) void k_rms(const float* __restrict__ x,
                                             const float* __restrict__ w,
                                             bf16_t* __restrict__ xn) {
  __shared__ float red[256];
  int row = blockIdx.x, tid = threadIdx.x;
  const float* xr = x + (size_t)row * DD;
  float ss = 0.f;
  for (int d = tid; d < DD; d += 256) { float v = xr[d]; ss += v * v; }
  red[tid] = ss; __syncthreads();
  for (int o = 128; o > 0; o >>= 1) { if (tid < o) red[tid] += red[tid + o]; __syncthreads(); }
  float scale = rsqrtf(red[0] * (1.0f / DD) + 1e-6f);
  for (int d = tid; d < DD; d += 256)
    xn[(size_t)row * DD + d] = f2bf(w[d] * xr[d] * scale);
}

// ---------------------------------------------------------------------------
// text_agg = mean over L of x_norm  -> [B, D] fp32
// ---------------------------------------------------------------------------
__global__ __launch_bounds__(256) void k_agg(const bf16_t* __restrict__ xn,
                                             float* __restrict__ agg) {
  int b = blockIdx.x / 10, dc = blockIdx.x % 10;
  int d = dc * 256 + threadIdx.x;
  float s = 0.f;
  for (int l = 0; l < LL; ++l)
    s += bf2f(xn[((size_t)(b * LL + l)) * DD + d]);
  agg[b * DD + d] = s * (1.0f / LL);
}

// ---------------------------------------------------------------------------
// Gate MLP: hidden = relu(agg @ fc1^T + b1); g = sigmoid(fc2.hidden + b2 + tgb)
// gatebuf[0..3] = per-batch task gate, gatebuf[4] = sigmoid(final_gate)
// ---------------------------------------------------------------------------
__global__ __launch_bounds__(256) void k_gate(const float* __restrict__ agg,
                                              const float* __restrict__ fc1,
                                              const float* __restrict__ fc1b,
                                              const float* __restrict__ fc2,
                                              const float* __restrict__ fc2b,
                                              const float* __restrict__ tgb,
                                              const float* __restrict__ fg,
                                              float* __restrict__ gatebuf) {
  __shared__ float hid[640];
  __shared__ float red[256];
  int b = blockIdx.x, tid = threadIdx.x;
  const float* ag = agg + b * DD;
  for (int j = tid; j < 640; j += 256) {
    float s = fc1b[j];
    const float* wr = fc1 + (size_t)j * DD;
    for (int d = 0; d < DD; ++d) s += wr[d] * ag[d];
    hid[j] = fmaxf(s, 0.f);
  }
  __syncthreads();
  float p = 0.f;
  for (int j = tid; j < 640; j += 256) p += fc2[j] * hid[j];
  red[tid] = p; __syncthreads();
  for (int o = 128; o > 0; o >>= 1) { if (tid < o) red[tid] += red[tid + o]; __syncthreads(); }
  if (tid == 0) {
    float z = red[0] + fc2b[0] + tgb[0];
    gatebuf[b] = 1.0f / (1.0f + __expf(-z));
    if (b == 0) gatebuf[4] = 1.0f / (1.0f + __expf(-fg[0]));
  }
}

// ---------------------------------------------------------------------------
// LoRA down: tmp[M,64] = X[M,2560] @ A^T.  8 waves/block, 2 M-tiles per wave
// so every B-fragment load feeds two WMMAs; all loads of a k-step issue as
// one clause ahead of the 8-WMMA burst.
// ---------------------------------------------------------------------------
__global__ __launch_bounds__(256) void k_lora_down(const bf16_t* __restrict__ X,
                                                   const bf16_t* __restrict__ A,
                                                   bf16_t* __restrict__ tmp) {
  int lane = threadIdx.x & 31, wave = threadIdx.x >> 5;
  int mbase = (blockIdx.x * 8 + wave) * 32;     // 2 M-tiles per wave
  v8f acc[2][4];
  for (int m = 0; m < 2; ++m)
    for (int nt = 0; nt < 4; ++nt)
      acc[m][nt] = (v8f){0.f,0.f,0.f,0.f,0.f,0.f,0.f,0.f};
  for (int kb = 0; kb < DD; kb += 32) {
    v16bf a0 = ld_a(X + (size_t)mbase * DD + kb, DD, lane);
    v16bf a1 = ld_a(X + (size_t)(mbase + 16) * DD + kb, DD, lane);
    v16bf b0 = ld_b(A + (size_t)(0 * 16) * DD + kb, DD, lane);
    v16bf b1 = ld_b(A + (size_t)(1 * 16) * DD + kb, DD, lane);
    v16bf b2 = ld_b(A + (size_t)(2 * 16) * DD + kb, DD, lane);
    v16bf b3 = ld_b(A + (size_t)(3 * 16) * DD + kb, DD, lane);
    acc[0][0] = wmma_bf(a0, b0, acc[0][0]);
    acc[1][0] = wmma_bf(a1, b0, acc[1][0]);
    acc[0][1] = wmma_bf(a0, b1, acc[0][1]);
    acc[1][1] = wmma_bf(a1, b1, acc[1][1]);
    acc[0][2] = wmma_bf(a0, b2, acc[0][2]);
    acc[1][2] = wmma_bf(a1, b2, acc[1][2]);
    acc[0][3] = wmma_bf(a0, b3, acc[0][3]);
    acc[1][3] = wmma_bf(a1, b3, acc[1][3]);
  }
  int hi8 = (lane >> 4) << 3;
  for (int m = 0; m < 2; ++m)
    for (int nt = 0; nt < 4; ++nt)
      for (int v = 0; v < 8; ++v)
        tmp[(size_t)(mbase + m * 16 + v + hi8) * RR + nt * 16 + (lane & 15)] =
            f2bf(acc[m][nt][v]);
}

// ---------------------------------------------------------------------------
// KV build (visual side): KV[M,2560] = tmp_vis[M,64] @ Bw^T, stored in two
// layouts: KV[b][n][d] (token-major, for QK^T) and KVt[b][h][hd][n] (d-major,
// for P.V).  A-fragments are invariant across all 160 n-tiles -> hoisted.
// ---------------------------------------------------------------------------
__global__ __launch_bounds__(256) void k_kv(const bf16_t* __restrict__ tmpv,
                                            const bf16_t* __restrict__ Bw,
                                            bf16_t* __restrict__ KV,
                                            bf16_t* __restrict__ KVt) {
  int lane = threadIdx.x & 31, wave = threadIdx.x >> 5;
  int mbase = (blockIdx.x * 8 + wave) * 16;
  int hi8 = (lane >> 4) << 3;
  v16bf a0 = ld_a(tmpv + (size_t)mbase * RR + 0, RR, lane);
  v16bf a1 = ld_a(tmpv + (size_t)mbase * RR + 32, RR, lane);
  for (int nt = 0; nt < 160; ++nt) {
    v16bf bf0 = ld_b(Bw + (size_t)(nt * 16) * RR + 0, RR, lane);
    v16bf bf1 = ld_b(Bw + (size_t)(nt * 16) * RR + 32, RR, lane);
    v8f acc = (v8f){0.f,0.f,0.f,0.f,0.f,0.f,0.f,0.f};
    acc = wmma_bf(a0, bf0, acc);
    acc = wmma_bf(a1, bf1, acc);
    int dcol = nt * 16 + (lane & 15);
    int h = dcol / HD, hd = dcol % HD;
    for (int v = 0; v < 8; ++v) {
      int grow = mbase + v + hi8;
      int b = grow >> 8, n = grow & 255;
      bf16_t val = f2bf(acc[v]);
      KV[((size_t)(b * NN + n)) * DD + dcol] = val;
      KVt[(((size_t)(b * HH + h)) * HD + hd) * NN + n] = val;
    }
  }
}

// ---------------------------------------------------------------------------
// Fused dual-branch attention + combine.
// Block = 256 thr (8 waves) handles (b, h, 32 L-rows).  Per-wave job mapping
// j = wave + 8*i  =>  mt = wave&1 (constant per wave), so each A-fragment is
// loaded once per k-step and reused across all of the wave's B-tiles.
// ---------------------------------------------------------------------------
__global__ __launch_bounds__(256) void k_attn(
    const bf16_t* __restrict__ tmp_s_txt, const bf16_t* __restrict__ tmp_t_txt,
    const bf16_t* __restrict__ Bs_bf, const bf16_t* __restrict__ Bt_bf,
    const bf16_t* __restrict__ KV_s, const bf16_t* __restrict__ KV_t,
    const bf16_t* __restrict__ KVt_s, const bf16_t* __restrict__ KVt_t,
    const unsigned char* __restrict__ mask, const float* __restrict__ text,
    const float* __restrict__ gatebuf, float* __restrict__ out) {
  __shared__ __align__(16) bf16_t Qs[32 * HD];          // 20 KB
  __shared__ __align__(16) bf16_t chunkB[128 * HD];     // 80 KB (K chunk / Vt chunk)
  __shared__ __align__(16) float  scoresS[32 * NN];     // 32 KB
  __shared__ __align__(16) bf16_t probsS[32 * NN];      // 16 KB
  __shared__ __align__(16) bf16_t tmptS[32 * RR];       // 4 KB
  __shared__ float redS[256];
  __shared__ float rowstat[32];

  int tid = threadIdx.x, lane = tid & 31, wave = tid >> 5;
  int bhz = blockIdx.x;
  int lt = bhz & 63, h = (bhz >> 6) & 7, b = bhz >> 9;
  int l0 = lt * 32;
  int hi8 = (lane >> 4) << 3;
  int mt = wave & 1;          // M-tile owned by this wave (all phases)
  int w2 = wave >> 1;         // N/D tile group base
  float g  = gatebuf[b];
  float fs = gatebuf[4];
  const float scale = 0.05590169943749474f;  // 320^-0.5

  v8f accO[2][5];

  for (int br = 0; br < 2; ++br) {
    const bf16_t* tmpTxt = br ? tmp_t_txt : tmp_s_txt;
    const bf16_t* Bbf    = br ? Bt_bf    : Bs_bf;
    const bf16_t* KV     = br ? KV_t     : KV_s;
    const bf16_t* KVt    = br ? KVt_t    : KVt_s;

    // stage tmp tile [32,64]
    {
      int row = tid >> 3, c8 = (tid & 7) * 8;
      *(U128*)&tmptS[row * RR + c8] =
          *(const U128*)(tmpTxt + (size_t)(b * LL + l0 + row) * RR + c8);
    }
    __syncthreads();

    // Q tile on the fly: Q = tmp @ Bw^T (head columns only).
    // A-frags (2 k-steps, fixed mt) loaded once, reused across 5 d-tiles.
    {
      v16bf qa0 = ld_a(tmptS + mt * 16 * RR + 0, RR, lane);
      v16bf qa1 = ld_a(tmptS + mt * 16 * RR + 32, RR, lane);
      for (int i = 0; i < 5; ++i) {
        int dt = w2 + 4 * i;
        v16bf bf0 = ld_b(Bbf + (size_t)(h * HD + dt * 16) * RR + 0, RR, lane);
        v16bf bf1 = ld_b(Bbf + (size_t)(h * HD + dt * 16) * RR + 32, RR, lane);
        v8f acc = (v8f){0.f,0.f,0.f,0.f,0.f,0.f,0.f,0.f};
        acc = wmma_bf(qa0, bf0, acc);
        acc = wmma_bf(qa1, bf1, acc);
        for (int v = 0; v < 8; ++v)
          Qs[(mt * 16 + v + hi8) * HD + dt * 16 + (lane & 15)] = f2bf(acc[v]);
      }
    }
    __syncthreads();

    // scores = Q K^T * scale (chunks of 128 tokens).  Per k-step: one A-frag
    // load feeds both of the wave's token tiles.
    for (int c = 0; c < 2; ++c) {
      for (int i = 0; i < 20; ++i) {   // 128 rows x 40 16B-chunks, async -> LDS
        int cid = tid + 256 * i;
        int tok = cid / 40, o = (cid % 40) * 8;
        async_copy_b128(&chunkB[tok * HD + o],
                        KV + ((size_t)(b * NN + c * 128 + tok)) * DD + h * HD + o);
      }
      wait_async();
      __syncthreads();
      {
        int nt0 = w2, nt1 = w2 + 4;
        v8f acc0 = (v8f){0.f,0.f,0.f,0.f,0.f,0.f,0.f,0.f};
        v8f acc1 = (v8f){0.f,0.f,0.f,0.f,0.f,0.f,0.f,0.f};
        for (int kb = 0; kb < HD; kb += 32) {
          v16bf a  = ld_a(Qs + mt * 16 * HD + kb, HD, lane);
          v16bf b0 = ld_b(chunkB + nt0 * 16 * HD + kb, HD, lane);
          v16bf b1 = ld_b(chunkB + nt1 * 16 * HD + kb, HD, lane);
          acc0 = wmma_bf(a, b0, acc0);
          acc1 = wmma_bf(a, b1, acc1);
        }
        for (int v = 0; v < 8; ++v) {
          int r = (mt * 16 + v + hi8) * NN + c * 128 + (lane & 15);
          scoresS[r + nt0 * 16] = acc0[v] * scale;
          scoresS[r + nt1 * 16] = acc1[v] * scale;
        }
      }
      __syncthreads();
    }

    // softmax over 256 tokens; thread = (row=lane, colgroup=wave)
    {
      int row = lane, sub = wave;
      float mx = -3.0e38f;
      for (int k = 0; k < 32; ++k) {
        int col = sub * 32 + k;
        float s = scoresS[row * NN + col];
        if (!mask[b * NN + col]) s = -3.0e38f;
        scoresS[row * NN + col] = s;
        mx = fmaxf(mx, s);
      }
      redS[sub * 32 + row] = mx;
      __syncthreads();
      if (wave == 0) {
        float m2 = redS[row];
        for (int s2 = 1; s2 < 8; ++s2) m2 = fmaxf(m2, redS[s2 * 32 + row]);
        rowstat[row] = m2;
      }
      __syncthreads();
      float rm = rowstat[row], sum = 0.f;
      for (int k = 0; k < 32; ++k) {
        int col = sub * 32 + k;
        float e = __expf(scoresS[row * NN + col] - rm);
        scoresS[row * NN + col] = e;
        sum += e;
      }
      redS[sub * 32 + row] = sum;
      __syncthreads();
      if (wave == 0) {
        float t = 0.f;
        for (int s2 = 0; s2 < 8; ++s2) t += redS[s2 * 32 + row];
        rowstat[row] = 1.0f / t;
      }
      __syncthreads();
      float inv = rowstat[row];
      for (int k = 0; k < 32; ++k) {
        int col = sub * 32 + k;
        probsS[row * NN + col] = f2bf(scoresS[row * NN + col] * inv);
      }
      __syncthreads();
    }

    // O = P @ V using d-major Vt chunks.  The wave's 4 probs A-frags per
    // chunk are loaded once and reused across its 5 d-tiles (20 WMMAs).
    for (int i = 0; i < 5; ++i)
      accO[br][i] = (v8f){0.f,0.f,0.f,0.f,0.f,0.f,0.f,0.f};
    for (int c = 0; c < 2; ++c) {
      for (int i = 0; i < 20; ++i) {   // 320 rows x 16 16B-chunks
        int cid = tid + 256 * i;
        int hd = cid >> 4, o = (cid & 15) * 8;
        async_copy_b128(&chunkB[hd * 128 + o],
                        KVt + (((size_t)(b * HH + h)) * HD + hd) * NN + c * 128 + o);
      }
      wait_async();
      __syncthreads();
      v16bf pa[4];
      for (int kbi = 0; kbi < 4; ++kbi)
        pa[kbi] = ld_a(probsS + mt * 16 * NN + c * 128 + kbi * 32, NN, lane);
      for (int i = 0; i < 5; ++i) {
        int dt = w2 + 4 * i;
        for (int kbi = 0; kbi < 4; ++kbi) {
          v16bf bb = ld_b(chunkB + dt * 16 * 128 + kbi * 32, 128, lane);
          accO[br][i] = wmma_bf(pa[kbi], bb, accO[br][i]);
        }
      }
      __syncthreads();
    }
  }

  // final combine: out = text + fs * (O_s + g * O_t)  (same job mapping)
  for (int i = 0; i < 5; ++i) {
    int dt = w2 + 4 * i;
    for (int v = 0; v < 8; ++v) {
      int row = l0 + mt * 16 + v + hi8;
      int col = h * HD + dt * 16 + (lane & 15);
      size_t idx = ((size_t)(b * LL + row)) * DD + col;
      out[idx] = text[idx] + fs * (accO[0][i][v] + g * accO[1][i][v]);
    }
  }
}

// ---------------------------------------------------------------------------
// Host: workspace layout + launch sequence
// ---------------------------------------------------------------------------
extern "C" void kernel_launch(void* const* d_in, const int* in_sizes, int n_in,
                              void* d_out, int out_size, void* d_ws, size_t ws_size,
                              hipStream_t stream) {
  const float* text = (const float*)d_in[0];
  const float* vis  = (const float*)d_in[1];
  const unsigned char* mask = (const unsigned char*)d_in[2];
  const float* A_s  = (const float*)d_in[3];
  const float* B_s  = (const float*)d_in[4];
  const float* A_t  = (const float*)d_in[5];
  const float* B_t  = (const float*)d_in[6];
  const float* fc1  = (const float*)d_in[7];
  const float* fc1b = (const float*)d_in[8];
  const float* fc2  = (const float*)d_in[9];
  const float* fc2b = (const float*)d_in[10];
  const float* tgb  = (const float*)d_in[11];
  const float* rmsw = (const float*)d_in[12];
  const float* fg   = (const float*)d_in[13];
  float* out = (float*)d_out;
  char* ws = (char*)d_ws;

  size_t o = 0;
  auto alloc = [&](size_t bytes) { size_t r = o; o += (bytes + 255) & ~(size_t)255; return r; };
  bf16_t* xnbf  = (bf16_t*)(ws + alloc((size_t)BB * LL * DD * 2));   // 41.9 MB
  bf16_t* visbf = (bf16_t*)(ws + alloc((size_t)BB * NN * DD * 2));   //  5.2 MB
  bf16_t* Asbf  = (bf16_t*)(ws + alloc((size_t)RR * DD * 2));
  bf16_t* Atbf  = (bf16_t*)(ws + alloc((size_t)RR * DD * 2));
  bf16_t* Bsbf  = (bf16_t*)(ws + alloc((size_t)DD * RR * 2));
  bf16_t* Btbf  = (bf16_t*)(ws + alloc((size_t)DD * RR * 2));
  bf16_t* tmpS  = (bf16_t*)(ws + alloc((size_t)BB * LL * RR * 2));
  bf16_t* tmpT  = (bf16_t*)(ws + alloc((size_t)BB * LL * RR * 2));
  bf16_t* tmpSv = (bf16_t*)(ws + alloc((size_t)BB * NN * RR * 2));
  bf16_t* tmpTv = (bf16_t*)(ws + alloc((size_t)BB * NN * RR * 2));
  bf16_t* KVs   = (bf16_t*)(ws + alloc((size_t)BB * NN * DD * 2));
  bf16_t* KVts  = (bf16_t*)(ws + alloc((size_t)BB * HH * HD * NN * 2));
  bf16_t* KVt_  = (bf16_t*)(ws + alloc((size_t)BB * NN * DD * 2));
  bf16_t* KVtt  = (bf16_t*)(ws + alloc((size_t)BB * HH * HD * NN * 2));
  float*  aggf  = (float*)(ws + alloc((size_t)BB * DD * 4));
  float*  gatef = (float*)(ws + alloc(64));
  (void)in_sizes; (void)n_in; (void)out_size; (void)ws_size;

  // fp32 -> bf16 weight/activation conversions
  int nv = BB * NN * DD, na = RR * DD, nb = DD * RR;
  k_cvt<<<(nv + 255) / 256, 256, 0, stream>>>(vis, visbf, nv);
  k_cvt<<<(na + 255) / 256, 256, 0, stream>>>(A_s, Asbf, na);
  k_cvt<<<(na + 255) / 256, 256, 0, stream>>>(A_t, Atbf, na);
  k_cvt<<<(nb + 255) / 256, 256, 0, stream>>>(B_s, Bsbf, nb);
  k_cvt<<<(nb + 255) / 256, 256, 0, stream>>>(B_t, Btbf, nb);

  k_rms<<<BB * LL, 256, 0, stream>>>(text, rmsw, xnbf);
  k_agg<<<BB * (DD / 256), 256, 0, stream>>>(xnbf, aggf);
  k_gate<<<BB, 256, 0, stream>>>(aggf, fc1, fc1b, fc2, fc2b, tgb, fg, gatef);

  // LoRA down (WMMA): text 8192 rows -> 512 Mtiles / (8 waves x 2) = 32 blocks
  k_lora_down<<<32, 256, 0, stream>>>(xnbf, Asbf, tmpS);
  k_lora_down<<<32, 256, 0, stream>>>(xnbf, Atbf, tmpT);
  k_lora_down<<<4, 256, 0, stream>>>(visbf, Asbf, tmpSv);
  k_lora_down<<<4, 256, 0, stream>>>(visbf, Atbf, tmpTv);

  // KV build (WMMA), dual layout
  k_kv<<<8, 256, 0, stream>>>(tmpSv, Bsbf, KVs, KVts);
  k_kv<<<8, 256, 0, stream>>>(tmpTv, Btbf, KVt_, KVtt);

  // fused dual-branch attention + combine: (B=4) x (H=8) x (L/32=64) blocks
  k_attn<<<BB * HH * (LL / 32), 256, 0, stream>>>(
      tmpS, tmpT, Bsbf, Btbf, KVs, KVt_, KVts, KVtt, mask, text, gatef, out);
}
